// MoE_44513041056360
// MI455X (gfx1250) — matrix operand order repaired and checked by reading, not compiled, not run on hardware
//
#include <hip/hip_runtime.h>
#include <hip/hip_bf16.h>

// ---------------------------------------------------------------------------
// DeepSeek-style MoE for MI455X (gfx1250, wave32, WMMA).
// Sparse dispatch (top-4 of 16 experts) + shared expert, bf16 WMMA w/ fp32 acc.
// Per-wave 32(M)x16(N) tiles: each B fragment (fp32 weights -> bf16 cvt) is
// reused by two WMMAs, halving weight bytes + cvt VALU per matrix op.
// ---------------------------------------------------------------------------

#define T_TOK 4096
#define HDIM  2048
#define NEXP  16
#define KSEL  4
#define IDIM  1408
#define SIDIM 2816

typedef __attribute__((ext_vector_type(8)))  __bf16 v8bf;
typedef __attribute__((ext_vector_type(16))) __bf16 v16bf;
typedef __attribute__((ext_vector_type(8)))  float  v8f;

// ---- fragment loaders (contiguous k order, same for A and B) ----------------------

// A-matrix 16x32 bf16: lane<16 -> row=lane, K=kk..kk+15;
//                      lane>=16 -> row=lane-16, K=kk+16..kk+31.
__device__ __forceinline__ v16bf load_a_bf16(const __bf16* __restrict__ rowp,
                                             int kk, int half) {
  const v8bf* p = reinterpret_cast<const v8bf*>(rowp + kk + (half << 4));
  v8bf a0 = p[0];   // K offset +0 (or +16)
  v8bf a1 = p[1];   // K offset +8 (or +24)
  v16bf r;
#pragma unroll
  for (int i = 0; i < 8; ++i) { r[i] = a0[i]; r[i + 8] = a1[i]; }
  return r;
}

// B-matrix 32x16 (KxN): lane<16 -> col=lane, K=kk..kk+15;
//                       lane>=16 -> col=lane-16, K=kk+16..kk+31.
// Weight row (output channel) is contiguous over K -> 4x float4 + cvt to bf16.
__device__ __forceinline__ v16bf load_b_f32(const float* __restrict__ wrow,
                                            int kk, int half) {
  const float4* q = reinterpret_cast<const float4*>(wrow + kk + (half << 4));
  float4 b0 = q[0], b1 = q[1], b2 = q[2], b3 = q[3];
  v16bf r;
  r[0]  = (__bf16)b0.x; r[1]  = (__bf16)b0.y; r[2]  = (__bf16)b0.z; r[3]  = (__bf16)b0.w;
  r[4]  = (__bf16)b1.x; r[5]  = (__bf16)b1.y; r[6]  = (__bf16)b1.z; r[7]  = (__bf16)b1.w;
  r[8]  = (__bf16)b2.x; r[9]  = (__bf16)b2.y; r[10] = (__bf16)b2.z; r[11] = (__bf16)b2.w;
  r[12] = (__bf16)b3.x; r[13] = (__bf16)b3.y; r[14] = (__bf16)b3.z; r[15] = (__bf16)b3.w;
  return r;
}

__device__ __forceinline__ float fast_silu_mul(float g, float u) {
  // silu(g)*u with v_rcp_f32 instead of the IEEE divide expansion
  float sig = __builtin_amdgcn_rcpf(1.f + __expf(-g));
  return g * sig * u;
}

// ---- small utility kernels ------------------------------------------------

__global__ void zero_counts_kernel(int* __restrict__ counts) {
  if (threadIdx.x < NEXP) counts[threadIdx.x] = 0;
}

__global__ void prefix_kernel(const int* __restrict__ counts,
                              int* __restrict__ offsets) {
  if (threadIdx.x == 0) {
    int s = 0;
    for (int e = 0; e < NEXP; ++e) { offsets[e] = s; s += counts[e]; }
    offsets[NEXP] = s;
  }
}

__global__ __launch_bounds__(256)
void cvt_bf16_kernel(const float* __restrict__ in, __bf16* __restrict__ out,
                     long n) {
  long i = ((long)blockIdx.x * blockDim.x + threadIdx.x) * 4;
  if (i + 3 < n) {
    float4 v = *reinterpret_cast<const float4*>(in + i);
    out[i + 0] = (__bf16)v.x;
    out[i + 1] = (__bf16)v.y;
    out[i + 2] = (__bf16)v.z;
    out[i + 3] = (__bf16)v.w;
  }
}

// ---- router: softmax over 16 logits, top-4, renormalize, slot assignment --

__global__ __launch_bounds__(128)
void router_kernel(const float* __restrict__ x, const float* __restrict__ gw,
                   int* __restrict__ counts, int* __restrict__ slot_token,
                   float* __restrict__ slot_w) {
  int wave = threadIdx.x >> 5;
  int lane = threadIdx.x & 31;
  int t = blockIdx.x * 4 + wave;
  if (t >= T_TOK) return;
  const float* xr = x + (size_t)t * HDIM;
  float acc[NEXP];
#pragma unroll
  for (int e = 0; e < NEXP; ++e) acc[e] = 0.f;
  for (int h = lane; h < HDIM; h += 32) {
    float xv = xr[h];
#pragma unroll
    for (int e = 0; e < NEXP; ++e) acc[e] += xv * gw[e * HDIM + h];
  }
#pragma unroll
  for (int e = 0; e < NEXP; ++e) {
    float v = acc[e];
#pragma unroll
    for (int off = 16; off; off >>= 1) v += __shfl_xor(v, off, 32);
    acc[e] = v;
  }
  if (lane == 0) {
    // top-4 by logit (softmax monotonic; renormalized top-k softmax ==
    // softmax over the top-k logits)
    int sel[KSEL]; float slog[KSEL];
    unsigned used = 0;
    for (int k = 0; k < KSEL; ++k) {
      float best = -3.4e38f; int bi = 0;
      for (int e = 0; e < NEXP; ++e)
        if (!((used >> e) & 1u) && acc[e] > best) { best = acc[e]; bi = e; }
      used |= 1u << bi; sel[k] = bi; slog[k] = best;
    }
    float m = slog[0], s = 0.f, ew[KSEL];
    for (int k = 0; k < KSEL; ++k) { ew[k] = __expf(slog[k] - m); s += ew[k]; }
    float inv = 1.f / s;
    for (int k = 0; k < KSEL; ++k) {
      int e = sel[k];
      int pos = atomicAdd(&counts[e], 1);
      slot_token[e * T_TOK + pos] = t;
      slot_w[e * T_TOK + pos] = ew[k] * inv;
    }
  }
}

// ---- GEMM1: S = silu(X Wg^T) * (X Wu^T), bf16 out -------------------------
// Block 256 = 8 waves (2 M x 4 N); per-wave 32x16 tile -> each B fragment
// feeds two WMMAs. 4 accumulators (g/u x 2 M-subtiles).

template <bool ROUTED>
__global__ __launch_bounds__(256)
void gemm1_swiglu(const __bf16* __restrict__ xb,
                  const float* __restrict__ Wg, const float* __restrict__ Wu,
                  int Ncols,
                  const int* __restrict__ counts,
                  const int* __restrict__ offsets,
                  const int* __restrict__ slot_token,
                  __bf16* __restrict__ S) {
  int e = blockIdx.z;
  int nrows = ROUTED ? counts[e] : T_TOK;
  int rowBlock = blockIdx.y * 64;
  if (rowBlock >= nrows) return;

  int wave = threadIdx.x >> 5;
  int lane = threadIdx.x & 31;
  int wm = wave & 1, wn = wave >> 1;
  int half = lane >> 4, l16 = lane & 15;
  int rowBase = rowBlock + wm * 32;
  int col = blockIdx.x * 64 + wn * 16 + l16;

  const float* wgr = Wg + (size_t)e * Ncols * HDIM + (size_t)col * HDIM;
  const float* wur = Wu + (size_t)e * Ncols * HDIM + (size_t)col * HDIM;

  // two A row pointers (M subtiles), clamped; garbage rows masked at store
  const __bf16* ap[2];
#pragma unroll
  for (int mt = 0; mt < 2; ++mt) {
    int arow = rowBase + mt * 16 + l16;
    int arc = arow < nrows ? arow : nrows - 1;
    if (ROUTED) {
      int t = slot_token[e * T_TOK + arc];
      ap[mt] = xb + (size_t)t * HDIM;
    } else {
      ap[mt] = xb + (size_t)arc * HDIM;
    }
  }

  v8f accg0 = {}, accg1 = {}, accu0 = {}, accu1 = {};
  for (int kk = 0; kk < HDIM; kk += 32) {
    v16bf a0 = load_a_bf16(ap[0], kk, half);
    v16bf a1 = load_a_bf16(ap[1], kk, half);
    v16bf bg = load_b_f32(wgr, kk, half);
    v16bf bu = load_b_f32(wur, kk, half);
    accg0 = __builtin_amdgcn_wmma_f32_16x16x32_bf16(false, a0, false, bg,
                                                    (short)0, accg0, false, false);
    accg1 = __builtin_amdgcn_wmma_f32_16x16x32_bf16(false, a1, false, bg,
                                                    (short)0, accg1, false, false);
    accu0 = __builtin_amdgcn_wmma_f32_16x16x32_bf16(false, a0, false, bu,
                                                    (short)0, accu0, false, false);
    accu1 = __builtin_amdgcn_wmma_f32_16x16x32_bf16(false, a1, false, bu,
                                                    (short)0, accu1, false, false);
  }

  size_t base = ROUTED ? (size_t)offsets[e] : (size_t)0;
#pragma unroll
  for (int mt = 0; mt < 2; ++mt) {
    const v8f& g8 = mt ? accg1 : accg0;
    const v8f& u8 = mt ? accu1 : accu0;
#pragma unroll
    for (int r = 0; r < 8; ++r) {
      int row = rowBase + mt * 16 + r + half * 8;  // D layout: rows r / r+8
      if (row < nrows) {
        float sv = fast_silu_mul(g8[r], u8[r]);
        S[(base + row) * (size_t)Ncols + col] = (__bf16)sv;
      }
    }
  }
}

// ---- GEMM2: down-projection; shared writes y, routed atomically adds ------

template <bool ROUTED>
__global__ __launch_bounds__(256)
void gemm2_down(const __bf16* __restrict__ S, int Kdim,
                const float* __restrict__ Wd,
                const int* __restrict__ counts,
                const int* __restrict__ offsets,
                const int* __restrict__ slot_token,
                const float* __restrict__ slot_w,
                float* __restrict__ y) {
  int e = blockIdx.z;
  int nrows = ROUTED ? counts[e] : T_TOK;
  int rowBlock = blockIdx.y * 64;
  if (rowBlock >= nrows) return;

  int wave = threadIdx.x >> 5;
  int lane = threadIdx.x & 31;
  int wm = wave & 1, wn = wave >> 1;
  int half = lane >> 4, l16 = lane & 15;
  int rowBase = rowBlock + wm * 32;
  int col = blockIdx.x * 64 + wn * 16 + l16;

  const float* wdr = Wd + (size_t)e * HDIM * Kdim + (size_t)col * Kdim;

  size_t base = ROUTED ? (size_t)offsets[e] : (size_t)0;
  const __bf16* ap[2];
#pragma unroll
  for (int mt = 0; mt < 2; ++mt) {
    int arow = rowBase + mt * 16 + l16;
    int arc = arow < nrows ? arow : nrows - 1;
    ap[mt] = S + (base + arc) * (size_t)Kdim;
  }

  v8f acc0 = {}, acc1 = {};
  for (int kk = 0; kk < Kdim; kk += 32) {
    v16bf a0 = load_a_bf16(ap[0], kk, half);
    v16bf a1 = load_a_bf16(ap[1], kk, half);
    v16bf b  = load_b_f32(wdr, kk, half);
    acc0 = __builtin_amdgcn_wmma_f32_16x16x32_bf16(false, a0, false, b,
                                                   (short)0, acc0, false, false);
    acc1 = __builtin_amdgcn_wmma_f32_16x16x32_bf16(false, a1, false, b,
                                                   (short)0, acc1, false, false);
  }

#pragma unroll
  for (int mt = 0; mt < 2; ++mt) {
    const v8f& a8 = mt ? acc1 : acc0;
#pragma unroll
    for (int r = 0; r < 8; ++r) {
      int row = rowBase + mt * 16 + r + half * 8;
      if (row < nrows) {
        if (ROUTED) {
          int   t = slot_token[e * T_TOK + row];
          float w = slot_w[e * T_TOK + row];
          atomicAdd(&y[(size_t)t * HDIM + col], w * a8[r]);
        } else {
          y[(size_t)row * HDIM + col] = a8[r];
        }
      }
    }
  }
}

// ---------------------------------------------------------------------------

extern "C" void kernel_launch(void* const* d_in, const int* in_sizes, int n_in,
                              void* d_out, int out_size, void* d_ws,
                              size_t ws_size, hipStream_t stream) {
  const float* x       = (const float*)d_in[0];  // [B,S,H]
  const float* gate_w  = (const float*)d_in[1];  // [E,H]
  const float* w_gate  = (const float*)d_in[2];  // [E,I,H]
  const float* w_up    = (const float*)d_in[3];  // [E,I,H]
  const float* w_down  = (const float*)d_in[4];  // [E,H,I]
  const float* sw_gate = (const float*)d_in[5];  // [SI,H]
  const float* sw_up   = (const float*)d_in[6];  // [SI,H]
  const float* sw_down = (const float*)d_in[7];  // [H,SI]
  float* y = (float*)d_out;

  // workspace carve-up (~87 MB total)
  char* p = (char*)d_ws;
  int*    counts     = (int*)p;      p += 256;
  int*    offsets    = (int*)p;      p += 256;
  int*    slot_token = (int*)p;      p += sizeof(int) * NEXP * T_TOK;
  float*  slot_w     = (float*)p;    p += sizeof(float) * NEXP * T_TOK;
  __bf16* xb         = (__bf16*)p;   p += sizeof(__bf16) * (size_t)T_TOK * HDIM;
  __bf16* Smoe       = (__bf16*)p;   p += sizeof(__bf16) * (size_t)T_TOK * KSEL * IDIM;
  __bf16* Ssh        = (__bf16*)p;   p += sizeof(__bf16) * (size_t)T_TOK * SIDIM;

  zero_counts_kernel<<<1, 32, 0, stream>>>(counts);

  long nx = (long)T_TOK * HDIM;
  cvt_bf16_kernel<<<(unsigned)((nx / 4 + 255) / 256), 256, 0, stream>>>(x, xb, nx);

  router_kernel<<<T_TOK / 4, 128, 0, stream>>>(x, gate_w, counts, slot_token,
                                               slot_w);

  prefix_kernel<<<1, 32, 0, stream>>>(counts, offsets);

  // routed experts: fused gate/up + SiLU -> compact bf16 intermediate
  dim3 g1r(IDIM / 64, T_TOK / 64, NEXP);
  gemm1_swiglu<true><<<g1r, 256, 0, stream>>>(xb, w_gate, w_up, IDIM, counts,
                                              offsets, slot_token, Smoe);
  // shared expert: same kernel, identity routing
  dim3 g1s(SIDIM / 64, T_TOK / 64, 1);
  gemm1_swiglu<false><<<g1s, 256, 0, stream>>>(xb, sw_gate, sw_up, SIDIM,
                                               nullptr, nullptr, nullptr, Ssh);

  // shared down-proj writes y (initializes every element) ...
  dim3 g2s(HDIM / 64, T_TOK / 64, 1);
  gemm2_down<false><<<g2s, 256, 0, stream>>>(Ssh, SIDIM, sw_down, nullptr,
                                             nullptr, nullptr, nullptr, y);
  // ... then routed down-proj scales by combine weight and atomically adds
  dim3 g2r(HDIM / 64, T_TOK / 64, NEXP);
  gemm2_down<true><<<g2r, 256, 0, stream>>>(Smoe, IDIM, w_down, counts,
                                            offsets, slot_token, slot_w, y);
}
